// TransformerBlock_85117661872380
// MI455X (gfx1250) — matrix operand order repaired and checked by reading, not compiled, not run on hardware
//
#include <hip/hip_runtime.h>
#include <hip/hip_bf16.h>

typedef __bf16 v16bf __attribute__((ext_vector_type(16)));
typedef __bf16 v8bf  __attribute__((ext_vector_type(8)));
typedef float  v8f   __attribute__((ext_vector_type(8)));
typedef unsigned u32x4 __attribute__((ext_vector_type(4)));
typedef int      i32x4 __attribute__((ext_vector_type(4)));
typedef int      i32x8 __attribute__((ext_vector_type(8)));

#define DEV_INLINE __device__ __forceinline__

// ---------------------------------------------------------------------------
// f32 -> bf16 conversion (weights), grid-stride
// ---------------------------------------------------------------------------
__global__ void cvt_f32_bf16(const float* __restrict__ in, __bf16* __restrict__ out, int n) {
    for (int i = blockIdx.x * blockDim.x + threadIdx.x; i < n; i += gridDim.x * blockDim.x)
        out[i] = (__bf16)in[i];
}

// ---------------------------------------------------------------------------
// LayerNorm: one block per row (D=1024, 256 threads, 4 elems/thread), bf16 out
// ---------------------------------------------------------------------------
__global__ void __launch_bounds__(256) ln_bf16(const float* __restrict__ x,
                                               const float* __restrict__ sc,
                                               const float* __restrict__ bi,
                                               __bf16* __restrict__ out) {
    const int row = blockIdx.x;
    const int t = threadIdx.x;
    const float* xr = x + (size_t)row * 1024;
    float vals[4], s = 0.f, s2 = 0.f;
#pragma unroll
    for (int i = 0; i < 4; ++i) {
        float v = xr[t + i * 256];
        vals[i] = v; s += v; s2 += v * v;
    }
#pragma unroll
    for (int m = 16; m >= 1; m >>= 1) { s += __shfl_xor(s, m); s2 += __shfl_xor(s2, m); }
    __shared__ float rs[8], rs2[8];
    const int w = t >> 5, lane = t & 31;
    if (lane == 0) { rs[w] = s; rs2[w] = s2; }
    __syncthreads();
    if (t == 0) {
        float a = 0.f, b = 0.f;
#pragma unroll
        for (int i = 0; i < 8; ++i) { a += rs[i]; b += rs2[i]; }
        rs[0] = a; rs2[0] = b;
    }
    __syncthreads();
    const float mu  = rs[0] * (1.f / 1024.f);
    const float var = rs2[0] * (1.f / 1024.f) - mu * mu;
    const float inv = rsqrtf(var + 1e-6f);
#pragma unroll
    for (int i = 0; i < 4; ++i) {
        int c = t + i * 256;
        out[(size_t)row * 1024 + c] = (__bf16)((vals[i] - mu) * inv * sc[c] + bi[c]);
    }
}

// ---------------------------------------------------------------------------
// Fragment loaders (CDNA5 16-bit WMMA VGPR layouts, wave32)
//   A 16x32: lane m=lane%16, k = (e&7) + 16*(e>>3) + 8*(lane>>4)
//   B 32x16: lane n=lane%16, k = (e&7) + 16*(e>>3) + 8*(lane>>4)   (loaded from B^T rows)
// ---------------------------------------------------------------------------
DEV_INLINE v16bf frag_a(const __bf16* p /*row base + 8*hi*/) {
    v8bf lo = *(const v8bf*)p;
    v8bf hh = *(const v8bf*)(p + 16);
    v16bf f;
#pragma unroll
    for (int e = 0; e < 8; ++e) { f[e] = lo[e]; f[e + 8] = hh[e]; }
    return f;
}
DEV_INLINE v16bf frag_b(const __bf16* p /*row base + 8*hi*/) {
    v8bf lo = *(const v8bf*)p;
    v8bf hh = *(const v8bf*)(p + 16);
    v16bf f;
#pragma unroll
    for (int e = 0; e < 8; ++e) { f[e] = lo[e]; f[e + 8] = hh[e]; }
    return f;
}

// ---------------------------------------------------------------------------
// Tiled WMMA GEMM: C[M,N] = A_bf16[M,K] @ B_bf16[K,N] (+bias, epilogue MODE)
//  MODE 0: bf16 out      MODE 1: bf16 gelu(out)      MODE 2: f32 resid + out
//  Block 128x128, 8 waves (2x4), wave tile 64x32, BK=32.
//  A tile (128x32) is staged by the Tensor Data Mover (tensor_load_to_lds),
//  with D# LDS padding 16DW+4DW -> row stride 40 elems (matches frag loads).
//  B tile staged manually transposed (overlaps TDM), then s_wait_tensorcnt.
// ---------------------------------------------------------------------------
template <int MODE>
__global__ void __launch_bounds__(256) gemm_bf16_wmma(
    const __bf16* __restrict__ A, const __bf16* __restrict__ Bm,
    const float* __restrict__ bias, const float* __restrict__ resid,
    float* __restrict__ outf, __bf16* __restrict__ outb,
    int M, int N, int K) {
    __shared__ __bf16 As[128 * 40];   // [m][k], TDM-padded, stride 40 (16B aligned)
    __shared__ __bf16 Bts[128 * 40];  // B^T tile: [n][k]
    const int t = threadIdx.x;
    const int w = t >> 5, lane = t & 31, l16 = lane & 15, hi = lane >> 4;
    const int wm = w & 1, wn = w >> 1;          // 2 x 4 wave grid
    const int m0 = blockIdx.y * 128, n0 = blockIdx.x * 128;
    const unsigned ldsA = (unsigned)(size_t)(void*)As;  // LDS byte offset (addr[31:0])

    v8f acc[4][2];
#pragma unroll
    for (int a = 0; a < 4; ++a)
#pragma unroll
        for (int b = 0; b < 2; ++b) acc[a][b] = (v8f){};

    for (int k0 = 0; k0 < K; k0 += 32) {
        __syncthreads();
        // --- wave 0: issue TDM load of the 128x32 A tile into As ---
        if (w == 0) {
            unsigned long long ga =
                (unsigned long long)(size_t)(A + (size_t)m0 * K + k0);
            u32x4 g0;
            g0[0] = 1u;                                   // count=1, user mode
            g0[1] = ldsA;                                 // lds_addr
            g0[2] = (unsigned)(ga & 0xffffffffu);         // global_addr lo
            g0[3] = (unsigned)((ga >> 32) & 0x01ffffffu)  // global_addr hi
                    | 0x80000000u;                        // type=2 (image)
            i32x8 g1;
            g1[0] = (1 << 16)      // data_size = 2 bytes
                  | (1 << 20)      // pad_enable
                  | (3 << 22)      // pad_interval: 16 DWORDs
                  | (3 << 25);     // pad_amount:   4 DWORDs  -> stride 40 elems
            g1[1] = (int)((K & 0xffff) << 16);                 // tensor_dim0 lo16
            g1[2] = (int)(((unsigned)K >> 16) | (((unsigned)M & 0xffffu) << 16));
            g1[3] = (int)(((unsigned)M >> 16) | (32u << 16));  // dim1 hi | tile_dim0=32
            g1[4] = 128;                                       // tile_dim1=128
            g1[5] = K;                                         // tensor_dim0_stride lo
            g1[6] = 0; g1[7] = 0;
            i32x4 z4 = {};
#if __clang_major__ >= 23
            i32x8 z8 = {};
            __builtin_amdgcn_tensor_load_to_lds(g0, g1, z4, z4, z8, 0);
#else
            __builtin_amdgcn_tensor_load_to_lds(g0, g1, z4, z4, 0);
#endif
        }
        // --- all threads: stage B^T tile (overlaps the TDM transfer) ---
#pragma unroll
        for (int it = 0; it < 2; ++it) {
            int idx = t + it * 256;                  // 512 chunks of 8 bf16
            int kr = idx >> 4, nc = (idx & 15) << 3; // 16 chunks per k-row
            v8bf d = *(const v8bf*)(Bm + (size_t)(k0 + kr) * N + n0 + nc);
#pragma unroll
            for (int e = 0; e < 8; ++e) Bts[(nc + e) * 40 + kr] = d[e];
        }
        if (k0 + 32 < K)  // gfx1250 global_prefetch_b8 of next B K-rows
            __builtin_prefetch(Bm + (size_t)(k0 + 32 + (t >> 3)) * N + n0 + (t & 7) * 16, 0, 1);
        if (w == 0) __builtin_amdgcn_s_wait_tensorcnt(0);
        __syncthreads();

        v16bf af[4], bf[2];
#pragma unroll
        for (int tm = 0; tm < 4; ++tm)
            af[tm] = frag_a(As + (wm * 64 + tm * 16 + l16) * 40 + 8 * hi);
#pragma unroll
        for (int tn = 0; tn < 2; ++tn)
            bf[tn] = frag_b(Bts + (wn * 32 + tn * 16 + l16) * 40 + 8 * hi);
#pragma unroll
        for (int tm = 0; tm < 4; ++tm)
#pragma unroll
            for (int tn = 0; tn < 2; ++tn)
                acc[tm][tn] = __builtin_amdgcn_wmma_f32_16x16x32_bf16(
                    false, af[tm], false, bf[tn], (short)0, acc[tm][tn], false, false);
    }

    // epilogue: lane element i -> m = i + 8*hi within tile, n = l16
#pragma unroll
    for (int tm = 0; tm < 4; ++tm) {
#pragma unroll
        for (int tn = 0; tn < 2; ++tn) {
            const int n = n0 + wn * 32 + tn * 16 + l16;
            const float bv = bias[n];
#pragma unroll
            for (int i = 0; i < 8; ++i) {
                const int m = m0 + wm * 64 + tm * 16 + i + 8 * hi;
                const size_t off = (size_t)m * N + n;
                float v = acc[tm][tn][i] + bv;
                if (MODE == 0) {
                    outb[off] = (__bf16)v;
                } else if (MODE == 1) {
                    float g = 0.5f * v *
                        (1.f + tanhf(0.7978845608028654f * (v + 0.044715f * v * v * v)));
                    outb[off] = (__bf16)g;
                } else {
                    outf[off] = resid[off] + v;
                }
            }
        }
    }
}

// ---------------------------------------------------------------------------
// Flash attention (causal): grid (S/128, B*H), 8 waves; each wave owns 16
// query rows, loops key blocks of 64 with online softmax. head_dim = 64.
// Masking VALU only spent on diagonal key blocks.
// ---------------------------------------------------------------------------
__global__ void __launch_bounds__(256) attn_wmma(
    const __bf16* __restrict__ Q, const __bf16* __restrict__ Kc,
    const __bf16* __restrict__ V, __bf16* __restrict__ O,
    int S, int D, int H) {
    const int bx = blockIdx.x;
    const int b  = blockIdx.y / H;
    const int h  = blockIdx.y % H;
    const int t  = threadIdx.x;
    const int w  = t >> 5, lane = t & 31, l16 = lane & 15, hi = lane >> 4;

    __shared__ __bf16 ks[64 * 72];       // K tile as-is = B^T for Q·K^T
    __shared__ __bf16 vts[64 * 72];      // V^T tile: [hd][key]
    __shared__ __bf16 ps[8 * 16 * 72];   // per-wave prob tiles (A for P·V)

    const int qrow0 = bx * 128 + w * 16;

    // Q A-fragments (constant over key loop): 2 K-steps of 32 over hd=64
    v16bf aq[2];
    {
        const __bf16* qp = Q + (size_t)(b * S + qrow0 + l16) * D + h * 64;
#pragma unroll
        for (int kk = 0; kk < 2; ++kk) aq[kk] = frag_a(qp + kk * 32 + 8 * hi);
    }

    float mrun[8], lrun[8];
    v8f oacc[4];
#pragma unroll
    for (int i = 0; i < 8; ++i) { mrun[i] = -1e30f; lrun[i] = 0.f; }
#pragma unroll
    for (int ht = 0; ht < 4; ++ht) oacc[ht] = (v8f){};

    const int nkb = 2 * bx + 2;  // causal: keys up to (bx+1)*128
    for (int kb = 0; kb < nkb; ++kb) {
        __syncthreads();
        // stage K (direct) and V (transposed) tiles, 64x64 bf16 each
#pragma unroll
        for (int it = 0; it < 2; ++it) {
            int idx = t + it * 256;           // 512 chunks of 8
            int r = idx >> 3, c = (idx & 7) << 3;
            size_t g = (size_t)(b * S + kb * 64 + r) * D + h * 64 + c;
            *(v8bf*)(ks + r * 72 + c) = *(const v8bf*)(Kc + g);
            v8bf dv = *(const v8bf*)(V + g);
#pragma unroll
            for (int e = 0; e < 8; ++e) vts[(c + e) * 72 + r] = dv[e];
        }
        __syncthreads();

        // scores: 16 rows x 64 keys per wave
        v8f sc[4];
#pragma unroll
        for (int tn = 0; tn < 4; ++tn) {
            v8f a = (v8f){};
#pragma unroll
            for (int kk = 0; kk < 2; ++kk) {
                v16bf bfr = frag_b(ks + (tn * 16 + l16) * 72 + kk * 32 + 8 * hi);
                a = __builtin_amdgcn_wmma_f32_16x16x32_bf16(
                    false, aq[kk], false, bfr, (short)0, a, false, false);
            }
            sc[tn] = a;
        }

        // scale (1/sqrt(64)); causal mask only needed on diagonal blocks
        float rmax[8];
#pragma unroll
        for (int i = 0; i < 8; ++i) rmax[i] = -1e30f;
        if (kb * 64 + 63 > qrow0) {  // diagonal block: per-element mask
#pragma unroll
            for (int tn = 0; tn < 4; ++tn) {
                const int kg = kb * 64 + tn * 16 + l16;
#pragma unroll
                for (int i = 0; i < 8; ++i) {
                    const int qg = qrow0 + i + 8 * hi;
                    float v = sc[tn][i] * 0.125f;
                    if (kg > qg) v = -1e30f;
                    sc[tn][i] = v;
                    rmax[i] = fmaxf(rmax[i], v);
                }
            }
        } else {                     // strictly-past block: no masking
#pragma unroll
            for (int tn = 0; tn < 4; ++tn)
#pragma unroll
                for (int i = 0; i < 8; ++i) {
                    float v = sc[tn][i] * 0.125f;
                    sc[tn][i] = v;
                    rmax[i] = fmaxf(rmax[i], v);
                }
        }
#pragma unroll
        for (int i = 0; i < 8; ++i)
#pragma unroll
            for (int m = 1; m < 16; m <<= 1)
                rmax[i] = fmaxf(rmax[i], __shfl_xor(rmax[i], m));

        float fac[8], rsum[8];
#pragma unroll
        for (int i = 0; i < 8; ++i) {
            float nm = fmaxf(mrun[i], rmax[i]);
            fac[i] = __expf(mrun[i] - nm);
            mrun[i] = nm;
            rsum[i] = 0.f;
        }
#pragma unroll
        for (int tn = 0; tn < 4; ++tn)
#pragma unroll
            for (int i = 0; i < 8; ++i) {
                float p = __expf(sc[tn][i] - mrun[i]);
                sc[tn][i] = p;
                rsum[i] += p;
            }
#pragma unroll
        for (int i = 0; i < 8; ++i) {
#pragma unroll
            for (int m = 1; m < 16; m <<= 1) rsum[i] += __shfl_xor(rsum[i], m);
            lrun[i] = lrun[i] * fac[i] + rsum[i];
        }
#pragma unroll
        for (int ht = 0; ht < 4; ++ht)
#pragma unroll
            for (int i = 0; i < 8; ++i) oacc[ht][i] *= fac[i];

        // bounce probs through per-wave LDS to re-layout D->A (wave-local, DS in-order)
        __bf16* pw = ps + w * 16 * 72;
#pragma unroll
        for (int tn = 0; tn < 4; ++tn)
#pragma unroll
            for (int i = 0; i < 8; ++i)
                pw[(i + 8 * hi) * 72 + tn * 16 + l16] = (__bf16)sc[tn][i];

        // O += P(16x64) @ V(64x64)
#pragma unroll
        for (int kk = 0; kk < 2; ++kk) {
            v16bf ap = frag_a(pw + l16 * 72 + kk * 32 + 8 * hi);
#pragma unroll
            for (int ht = 0; ht < 4; ++ht) {
                v16bf bv = frag_b(vts + (ht * 16 + l16) * 72 + kk * 32 + 8 * hi);
                oacc[ht] = __builtin_amdgcn_wmma_f32_16x16x32_bf16(
                    false, ap, false, bv, (short)0, oacc[ht], false, false);
            }
        }
    }

    // final normalize + store bf16 (layout [b*S+s][h*64+hd])
#pragma unroll
    for (int ht = 0; ht < 4; ++ht)
#pragma unroll
        for (int i = 0; i < 8; ++i) {
            const int qg = qrow0 + i + 8 * hi;
            float v = oacc[ht][i] / lrun[i];
            O[(size_t)(b * S + qg) * D + h * 64 + ht * 16 + l16] = (__bf16)v;
        }
}

// ---------------------------------------------------------------------------
// Host orchestration
// ---------------------------------------------------------------------------
extern "C" void kernel_launch(void* const* d_in, const int* in_sizes, int n_in,
                              void* d_out, int out_size, void* d_ws, size_t ws_size,
                              hipStream_t stream) {
    (void)in_sizes; (void)n_in; (void)out_size; (void)ws_size;
    const float* x    = (const float*)d_in[0];
    // d_in[1] = mask (causal tril; implemented analytically)
    const float* ln1s = (const float*)d_in[2];
    const float* ln1b = (const float*)d_in[3];
    const float* Wq   = (const float*)d_in[4];
    const float* bq   = (const float*)d_in[5];
    const float* Wk   = (const float*)d_in[6];
    const float* bk   = (const float*)d_in[7];
    const float* Wv   = (const float*)d_in[8];
    const float* bv   = (const float*)d_in[9];
    const float* Wo   = (const float*)d_in[10];
    const float* bo   = (const float*)d_in[11];
    const float* ln2s = (const float*)d_in[12];
    const float* ln2b = (const float*)d_in[13];
    const float* W1   = (const float*)d_in[14];
    const float* b1   = (const float*)d_in[15];
    const float* W2   = (const float*)d_in[16];
    const float* b2   = (const float*)d_in[17];
    float* out = (float*)d_out;

    const int R = 4096;   // B*S
    const int Dm = 1024, Ff = 4096, Ss = 2048, Hh = 16;
    const size_t Mi = 1024ull * 1024ull;

    __bf16* ws = (__bf16*)d_ws;
    __bf16 *wqb = ws,          *wkb = ws + Mi,     *wvb = ws + 2 * Mi;
    __bf16 *wob = ws + 3 * Mi, *w1b = ws + 4 * Mi, *w2b = ws + 8 * Mi;
    __bf16 *xn  = ws + 12 * Mi;
    __bf16 *qb_ = ws + 16 * Mi, *kb_ = ws + 20 * Mi, *vb_ = ws + 24 * Mi;
    __bf16 *ob_ = ws + 28 * Mi, *h1_ = ws + 32 * Mi;  // h1: 16Mi elems

    // 1) weights -> bf16
    cvt_f32_bf16<<<dim3(2048), 256, 0, stream>>>(Wq, wqb, (int)Mi);
    cvt_f32_bf16<<<dim3(2048), 256, 0, stream>>>(Wk, wkb, (int)Mi);
    cvt_f32_bf16<<<dim3(2048), 256, 0, stream>>>(Wv, wvb, (int)Mi);
    cvt_f32_bf16<<<dim3(2048), 256, 0, stream>>>(Wo, wob, (int)Mi);
    cvt_f32_bf16<<<dim3(2048), 256, 0, stream>>>(W1, w1b, (int)(4 * Mi));
    cvt_f32_bf16<<<dim3(2048), 256, 0, stream>>>(W2, w2b, (int)(4 * Mi));

    // 2) LN1 -> bf16
    ln_bf16<<<dim3(R), 256, 0, stream>>>(x, ln1s, ln1b, xn);

    // 3) Q/K/V projections (bf16 out)
    gemm_bf16_wmma<0><<<dim3(8, 32), 256, 0, stream>>>(xn, wqb, bq, nullptr, nullptr, qb_, R, Dm, Dm);
    gemm_bf16_wmma<0><<<dim3(8, 32), 256, 0, stream>>>(xn, wkb, bk, nullptr, nullptr, kb_, R, Dm, Dm);
    gemm_bf16_wmma<0><<<dim3(8, 32), 256, 0, stream>>>(xn, wvb, bv, nullptr, nullptr, vb_, R, Dm, Dm);

    // 4) flash attention -> bf16 O
    attn_wmma<<<dim3(Ss / 128, 2 * Hh), 256, 0, stream>>>(qb_, kb_, vb_, ob_, Ss, Dm, Hh);

    // 5) x1 = x + O @ Wo + bo  (f32 into d_out)
    gemm_bf16_wmma<2><<<dim3(8, 32), 256, 0, stream>>>(ob_, wob, bo, x, out, nullptr, R, Dm, Dm);

    // 6) LN2 -> bf16 (reuse xn)
    ln_bf16<<<dim3(R), 256, 0, stream>>>(out, ln2s, ln2b, xn);

    // 7) h1 = gelu(xn @ W1 + b1) -> bf16
    gemm_bf16_wmma<1><<<dim3(32, 32), 256, 0, stream>>>(xn, w1b, b1, nullptr, nullptr, h1_, R, Ff, Dm);

    // 8) out = x1 + h1 @ W2 + b2 (reads resid=d_out, writes d_out; 1:1 element map)
    gemm_bf16_wmma<2><<<dim3(8, 32), 256, 0, stream>>>(h1_, w2b, b2, out, out, nullptr, R, Dm, Ff);
}
